// MultiscaleDeformableAttention_67671504715880
// MI455X (gfx1250) — compile-verified, matrix operand order checked by reading, not run on hardware
//
#include <hip/hip_runtime.h>
#include <hip/hip_bf16.h>

// ---- problem constants --------------------------------------------------
#define NBATCH   2
#define QLEN     8192
#define NQ       (NBATCH * QLEN)    // 16384 query rows
#define EMBED    256
#define MHEAD    8
#define DVDIM    32
#define S_TOTAL  21760              // 128^2 + 64^2 + 32^2 + 16^2
#define NS       (NBATCH * S_TOTAL) // 43520 value rows

typedef __attribute__((ext_vector_type(16))) __bf16 v16bf;
typedef __attribute__((ext_vector_type(8)))  float  v8f;

union FragU { v16bf v; unsigned u[8]; };

__device__ __forceinline__ unsigned pack_bf2(float a, float b) {
    union { __bf16 h[2]; unsigned u; } p;
    p.h[0] = (__bf16)a;              // hardware f32->bf16 convert
    p.h[1] = (__bf16)b;
    return p.u;
}
__device__ __forceinline__ unsigned short to_bf16(float a) {
    union { __bf16 h; unsigned short s; } p;
    p.h = (__bf16)a;
    return p.s;
}
__device__ __forceinline__ float sigmoidf(float x) {
    return 1.0f / (1.0f + __expf(-x));
}

// ---- weight transpose + bf16 convert: Wt[col][k] = bf16(W[k][col]) ------
__global__ void wprep_kernel(const float* __restrict__ W,
                             unsigned short* __restrict__ Wt,
                             int ncols, int total) {
    int idx = blockIdx.x * blockDim.x + threadIdx.x;
    if (idx >= total) return;
    int col = idx >> 8;        // / 256
    int k   = idx & 255;
    Wt[idx] = to_bf16(W[(size_t)k * ncols + col]);
}

// ---- bf16 WMMA GEMM: out[r][c] = (X@Wt^T)[r][c] + bias[c] (opt * mask[r])
// X:    rows x 256 fp32 (row major), converted to bf16 while staging to LDS
// WtU:  colsN x 128 dwords (= colsN x 256 bf16, K contiguous per out-col)
// block tile 128(M) x 128(N); 8 waves (4x2), each wave -> 32x64 = 2x4 WMMA
__global__ __launch_bounds__(256) void gemm_bf16_kernel(
    const float*    __restrict__ X,
    const unsigned* __restrict__ WtU,
    const float*    __restrict__ bias,
    const float*    __restrict__ mask,   // may be nullptr
    float*          __restrict__ out,
    int colsN)
{
    __shared__ unsigned ldsA[128 * 18];  // 128 rows x 32 bf16, stride 18 dw
    __shared__ unsigned ldsB[128 * 18];  // 128 cols x 32 bf16, stride 18 dw

    const int tid  = threadIdx.x;
    const int lane = tid & 31;
    const int wave = tid >> 5;
    const int wr   = (wave >> 1) * 32;   // wave row offset in block tile
    const int wc   = (wave & 1)  * 64;   // wave col offset in block tile
    const int r    = lane & 15;
    const int half = lane >> 4;
    const size_t rowBase = (size_t)blockIdx.y * 128;
    const int    colBase = blockIdx.x * 128;

    v8f zero = {0.f,0.f,0.f,0.f,0.f,0.f,0.f,0.f};
    v8f acc[2][4];
    #pragma unroll
    for (int ti = 0; ti < 2; ++ti)
        #pragma unroll
        for (int tj = 0; tj < 4; ++tj) acc[ti][tj] = zero;

    for (int k0 = 0; k0 < EMBED; k0 += 32) {
        // prefetch next K tile of X (global_prefetch_b8)
        if (k0 + 32 < EMBED)
            __builtin_prefetch(X + (rowBase + (tid >> 1)) * EMBED + k0 + 32, 0, 0);

        // stage A: 128 rows x 32 k, fp32 -> packed bf16x2
        #pragma unroll
        for (int i = tid; i < 128 * 16; i += 256) {
            int row = i >> 4, kd = i & 15;
            const float2 f =
                reinterpret_cast<const float2*>(X + (rowBase + row) * EMBED + k0)[kd];
            ldsA[row * 18 + kd] = pack_bf2(f.x, f.y);
        }
        // stage B: 128 out-cols x 32 k (already bf16, dword copy)
        #pragma unroll
        for (int i = tid; i < 128 * 16; i += 256) {
            int row = i >> 4, kd = i & 15;
            ldsB[row * 18 + kd] = WtU[(size_t)(colBase + row) * 128 + (k0 >> 1) + kd];
        }
        __syncthreads();

        // fragment loads per ISA 16-bit A layout:
        // lane -> row (M or N), lane[4] -> K half, VGPR v -> K pair
        FragU a[2], b[4];
        #pragma unroll
        for (int v = 0; v < 8; ++v) {
            int kd = ((v < 4) ? v : v + 4) + 4 * half;   // dword index of K pair
            #pragma unroll
            for (int ti = 0; ti < 2; ++ti)
                a[ti].u[v] = ldsA[(wr + ti * 16 + r) * 18 + kd];
            #pragma unroll
            for (int tj = 0; tj < 4; ++tj)
                b[tj].u[v] = ldsB[(wc + tj * 16 + r) * 18 + kd];
        }
        #pragma unroll
        for (int ti = 0; ti < 2; ++ti)
            #pragma unroll
            for (int tj = 0; tj < 4; ++tj)
                acc[ti][tj] = __builtin_amdgcn_wmma_f32_16x16x32_bf16(
                    false, a[ti].v, false, b[tj].v,
                    (short)0, acc[ti][tj], false, false);
        __syncthreads();
    }

    // epilogue: C layout — VGPR d, row = d + 8*half, col = lane&15
    #pragma unroll
    for (int ti = 0; ti < 2; ++ti) {
        #pragma unroll
        for (int tj = 0; tj < 4; ++tj) {
            #pragma unroll
            for (int d = 0; d < 8; ++d) {
                size_t row = rowBase + wr + ti * 16 + d + 8 * half;
                int    col = colBase + wc + tj * 16 + r;
                float  v   = acc[ti][tj][d] + bias[col];
                if (mask) v *= mask[row];
                out[row * (size_t)colsN + col] = v;
            }
        }
    }
}

// ---- softmax over groups of 16 (per query row, per head) ----------------
__global__ void softmax16_kernel(float* __restrict__ attn) {
    int g = blockIdx.x * blockDim.x + threadIdx.x;
    if (g >= NQ * MHEAD) return;
    float* p = attn + (size_t)(g >> 3) * 128 + (g & 7) * 16;
    float v[16], mx = -1e30f;
    #pragma unroll
    for (int i = 0; i < 16; ++i) { v[i] = p[i]; mx = fmaxf(mx, v[i]); }
    float s = 0.f;
    #pragma unroll
    for (int i = 0; i < 16; ++i) { v[i] = __expf(v[i] - mx); s += v[i]; }
    float inv = 1.0f / s;
    #pragma unroll
    for (int i = 0; i < 16; ++i) p[i] = v[i] * inv;
}

// ---- deltas -> clipped sampling grid, in place --------------------------
// pos = sigmoid(xy) + delta * sigmoid(wh) * (SCALE/K);  grid = clip(2p-1)
__global__ void pos_kernel(float* __restrict__ grid,
                           const float* __restrict__ geom) {
    int gid = blockIdx.x * blockDim.x + threadIdx.x;   // NQ*128 (x,y) pairs
    if (gid >= NQ * 128) return;
    int row = gid >> 7;
    int c   = gid & 127;                               // m*16 + lk
    const float* g = geom + (size_t)row * 4;
    float s0 = sigmoidf(g[0]), s1 = sigmoidf(g[1]);
    float s2 = sigmoidf(g[2]), s3 = sigmoidf(g[3]);
    float* d = grid + (size_t)row * 256 + c * 2;
    float px = s0 + d[0] * (s2 * 0.125f);              // SCALE/K = 0.5/4
    float py = s1 + d[1] * (s3 * 0.125f);
    d[0] = fminf(fmaxf(2.f * px - 1.f, -1.f), 1.f);
    d[1] = fminf(fmaxf(2.f * py - 1.f, -1.f), 1.f);
}

// ---- bilinear sampler: one wave per (query row, head), lane = DV channel
__device__ __forceinline__ float corner_val(const float* __restrict__ vm,
                                            size_t base, size_t colOff,
                                            int xi, int yi, int W, int H,
                                            float w) {
    bool valid = (xi >= 0) & (xi < W) & (yi >= 0) & (yi < H);
    int xc = min(max(xi, 0), W - 1);
    int yc = min(max(yi, 0), H - 1);
    float val = vm[(base + (size_t)yc * W + xc) * 256 + colOff];
    return valid ? w * val : 0.f;
}

__global__ __launch_bounds__(256) void sample_kernel(
    const float* __restrict__ vmask,   // NS x 256  (m-major head channels)
    const float* __restrict__ grid,    // NQ x 256
    const float* __restrict__ attn,    // NQ x 128
    float*       __restrict__ sfeat)   // NQ x 256
{
    const int unit = blockIdx.x * 8 + (threadIdx.x >> 5); // (row, head)
    const int lane = threadIdx.x & 31;                    // DV channel
    const int row  = unit >> 3;
    const int m    = unit & 7;
    const int n    = row / QLEN;
    const float* gr = grid + (size_t)row * 256 + m * 32;
    const float* at = attn + (size_t)row * 128 + m * 16;
    const int dims[4]   = {128, 64, 32, 16};
    const int starts[4] = {0, 16384, 20480, 21504};
    const size_t colOff = m * 32 + lane;

    float acc = 0.f;
    #pragma unroll
    for (int l = 0; l < 4; ++l) {
        const int W = dims[l], H = dims[l];
        const size_t base = (size_t)n * S_TOTAL + starts[l];
        #pragma unroll
        for (int k = 0; k < 4; ++k) {
            int   lk = l * 4 + k;
            float gx = gr[lk * 2], gy = gr[lk * 2 + 1];
            float a  = at[lk];
            float x  = (gx + 1.f) * (W * 0.5f) - 0.5f;
            float y  = (gy + 1.f) * (H * 0.5f) - 0.5f;
            float x0f = floorf(x), y0f = floorf(y);
            float wx = x - x0f, wy = y - y0f;
            int   x0 = (int)x0f, y0 = (int)y0f;
            float s = corner_val(vmask, base, colOff, x0,     y0,     W, H, (1.f-wx)*(1.f-wy))
                    + corner_val(vmask, base, colOff, x0 + 1, y0,     W, H, wx*(1.f-wy))
                    + corner_val(vmask, base, colOff, x0,     y0 + 1, W, H, (1.f-wx)*wy)
                    + corner_val(vmask, base, colOff, x0 + 1, y0 + 1, W, H, wx*wy);
            acc += a * s;
        }
    }
    sfeat[(size_t)row * 256 + m * 32 + lane] = acc;
}

// ---- driver -------------------------------------------------------------
extern "C" void kernel_launch(void* const* d_in, const int* in_sizes, int n_in,
                              void* d_out, int out_size, void* d_ws, size_t ws_size,
                              hipStream_t stream) {
    (void)in_sizes; (void)n_in; (void)out_size; (void)ws_size;

    const float* queries = (const float*)d_in[0];
    const float* geom    = (const float*)d_in[1];
    const float* value   = (const float*)d_in[2];
    const float* vmaskIn = (const float*)d_in[3];
    const float* W_off   = (const float*)d_in[4];
    const float* b_off   = (const float*)d_in[5];
    const float* W_attn  = (const float*)d_in[6];
    const float* b_attn  = (const float*)d_in[7];
    const float* W_val   = (const float*)d_in[8];
    const float* b_val   = (const float*)d_in[9];
    const float* W_out   = (const float*)d_in[10];
    const float* b_out   = (const float*)d_in[11];
    float* out = (float*)d_out;

    char* ws = (char*)d_ws;
    size_t off = 0;
    auto wsalloc = [&](size_t bytes) -> void* {
        void* p = ws + off;
        off += (bytes + 255) & ~(size_t)255;
        return p;
    };
    unsigned short* wt_attn = (unsigned short*)wsalloc((size_t)128 * 256 * 2);
    unsigned short* wt_off  = (unsigned short*)wsalloc((size_t)256 * 256 * 2);
    unsigned short* wt_val  = (unsigned short*)wsalloc((size_t)256 * 256 * 2);
    unsigned short* wt_out  = (unsigned short*)wsalloc((size_t)256 * 256 * 2);
    float* attnBuf = (float*)wsalloc((size_t)NQ * 128 * 4);
    float* gridBuf = (float*)wsalloc((size_t)NQ * 256 * 4);
    float* vmasked = (float*)wsalloc((size_t)NS * 256 * 4);
    float* sfeat   = (float*)wsalloc((size_t)NQ * 256 * 4);

    // 1) weight transposes to bf16 Wt[col][k]
    wprep_kernel<<<(128 * 256 + 255) / 256, 256, 0, stream>>>(W_attn, wt_attn, 128, 128 * 256);
    wprep_kernel<<<(256 * 256 + 255) / 256, 256, 0, stream>>>(W_off,  wt_off,  256, 256 * 256);
    wprep_kernel<<<(256 * 256 + 255) / 256, 256, 0, stream>>>(W_val,  wt_val,  256, 256 * 256);
    wprep_kernel<<<(256 * 256 + 255) / 256, 256, 0, stream>>>(W_out,  wt_out,  256, 256 * 256);

    // 2) v_masked = mask * (value @ W_val + b_val)   [43520 x 256]
    gemm_bf16_kernel<<<dim3(256 / 128, NS / 128), 256, 0, stream>>>(
        value, (const unsigned*)wt_val, b_val, vmaskIn, vmasked, 256);

    // 3) attention logits + softmax                  [16384 x 128]
    gemm_bf16_kernel<<<dim3(128 / 128, NQ / 128), 256, 0, stream>>>(
        queries, (const unsigned*)wt_attn, b_attn, nullptr, attnBuf, 128);
    softmax16_kernel<<<(NQ * MHEAD + 255) / 256, 256, 0, stream>>>(attnBuf);

    // 4) offsets -> clipped sampling grid            [16384 x 256]
    gemm_bf16_kernel<<<dim3(256 / 128, NQ / 128), 256, 0, stream>>>(
        queries, (const unsigned*)wt_off, b_off, nullptr, gridBuf, 256);
    pos_kernel<<<(NQ * 128 + 255) / 256, 256, 0, stream>>>(gridBuf, geom);

    // 5) bilinear sampling + attention-weighted sum  [16384 x 256]
    sample_kernel<<<NQ * MHEAD / 8, 256, 0, stream>>>(vmasked, gridBuf, attnBuf, sfeat);

    // 6) output projection into d_out                [16384 x 256]
    gemm_bf16_kernel<<<dim3(256 / 128, NQ / 128), 256, 0, stream>>>(
        sfeat, (const unsigned*)wt_out, b_out, nullptr, out, 256);
}
